// OriginalDecoderWrapper_6897717477741
// MI455X (gfx1250) — matrix operand order repaired and checked by reading, not compiled, not run on hardware
//
#include <hip/hip_runtime.h>
#include <hip/hip_bf16.h>

typedef __attribute__((ext_vector_type(16))) _Float16 v16h;
typedef __attribute__((ext_vector_type(8)))  float    v8f;
typedef unsigned int u32x4 __attribute__((ext_vector_type(4)));
typedef int          i32x8 __attribute__((ext_vector_type(8)));
typedef int          i32x4 __attribute__((ext_vector_type(4)));

#define B_    8
#define S_    512
#define D_    512
#define H_    8
#define DK_   64
#define DFF_  2048
#define L_    6
#define V_    32000
#define FLAT_ 3000

// ---------------------------------------------------------------------------
// f32 -> f16 bulk convert (weights, once per launch). n % 4 == 0.
// ---------------------------------------------------------------------------
__global__ __launch_bounds__(256)
void cvt_f16_kernel(const float* __restrict__ in, _Float16* __restrict__ out,
                    long long n4) {
  long long i = ((long long)blockIdx.x * 256 + threadIdx.x);
  if (i >= n4) return;
  float4 v = *(const float4*)&in[i * 4];
  out[i * 4 + 0] = (_Float16)v.x;
  out[i * 4 + 1] = (_Float16)v.y;
  out[i * 4 + 2] = (_Float16)v.z;
  out[i * 4 + 3] = (_Float16)v.w;
}

// ---------------------------------------------------------------------------
// GEMV: out[b,n] = bias[n] + sum_k vin[b,k]*W[n,k]   (fc + collapsed cross-attn)
// ---------------------------------------------------------------------------
__global__ __launch_bounds__(64)
void gemv_kernel(const float* __restrict__ vin, const float* __restrict__ W,
                 const float* __restrict__ bias, float* __restrict__ vout,
                 int Kd, int Nd) {
  int n = blockIdx.x * 64 + threadIdx.x;
  int b = blockIdx.y;
  if (n >= Nd) return;
  const float* vr = vin + (size_t)b * Kd;
  const float* wr = W   + (size_t)n * Kd;
  float acc = bias ? bias[n] : 0.0f;
  for (int k = 0; k < Kd; ++k) acc += vr[k] * wr[k];
  vout[(size_t)b * Nd + n] = acc;
}

// ---------------------------------------------------------------------------
// Embedding * sqrt(D) + sinusoidal positional encoding (x stays f32)
// ---------------------------------------------------------------------------
__global__ __launch_bounds__(256)
void embed_kernel(const int* __restrict__ stacks, const float* __restrict__ emb,
                  float* __restrict__ x) {
  size_t i = (size_t)blockIdx.x * 256 + threadIdx.x;
  if (i >= (size_t)B_ * S_ * D_) return;
  int d = (int)(i & (D_ - 1));
  size_t bs = i >> 9;
  int s = (int)(bs & (S_ - 1));
  int b = (int)(bs >> 9);
  int tok = (s == 0) ? 1 : stacks[b * S_ + s - 1];
  int pair = d >> 1;
  float div = __expf((float)pair * (-2.0f * 9.210340371976184f / (float)D_));
  float ang = (float)s * div;
  float pe = (d & 1) ? __cosf(ang) : __sinf(ang);
  x[i] = emb[(size_t)tok * D_ + d] * 22.62741699796952f + pe;
}

// ---------------------------------------------------------------------------
// LayerNorm (unbiased variance /(D-1)), f32 in, f16 out (feeds WMMA GEMMs)
// ---------------------------------------------------------------------------
__global__ __launch_bounds__(256)
void ln_kernel(const float* __restrict__ x, const float* __restrict__ ga,
               const float* __restrict__ gb, _Float16* __restrict__ out) {
  int row = blockIdx.x;
  const float* xr = x + (size_t)row * D_;
  int t = threadIdx.x;
  float v0 = xr[t], v1 = xr[t + 256];
  float s = v0 + v1, ss = v0 * v0 + v1 * v1;
  #pragma unroll
  for (int m = 1; m < 32; m <<= 1) {
    s  += __shfl_xor(s,  m, 32);
    ss += __shfl_xor(ss, m, 32);
  }
  __shared__ float rs[8], rss[8];
  int w = t >> 5, lane = t & 31;
  if (lane == 0) { rs[w] = s; rss[w] = ss; }
  __syncthreads();
  if (t == 0) {
    float a = 0.f, c = 0.f;
    #pragma unroll
    for (int i = 0; i < 8; ++i) { a += rs[i]; c += rss[i]; }
    rs[0] = a; rss[0] = c;
  }
  __syncthreads();
  s = rs[0]; ss = rss[0];
  float mean = s * (1.0f / D_);
  float var  = (ss - s * mean) * (1.0f / (D_ - 1));
  float inv  = rsqrtf(var + 1e-6f);
  _Float16* orow = out + (size_t)row * D_;
  orow[t]       = (_Float16)(ga[t]       * (v0 - mean) * inv + gb[t]);
  orow[t + 256] = (_Float16)(ga[t + 256] * (v1 - mean) * inv + gb[t + 256]);
}

// ---------------------------------------------------------------------------
// x[b,s,:] += v[b,:]   (cross-attention broadcast residual)
// ---------------------------------------------------------------------------
__global__ __launch_bounds__(256)
void addvec_kernel(float* __restrict__ x, const float* __restrict__ v) {
  size_t i = (size_t)blockIdx.x * 256 + threadIdx.x;
  if (i >= (size_t)B_ * S_ * D_) return;
  int d = (int)(i & (D_ - 1));
  int b = (int)(i >> 18);
  x[i] += v[b * D_ + d];
}

// ---------------------------------------------------------------------------
// TDM: DMA one 2D f16 tile (tile_w x tile_h, row stride rowStride elements)
// from global to LDS, with hardware row padding 16dw data + 2dw pad so rows
// land at 36-half stride (matches the LDS fragment layout below).
// D# packing per CDNA5 ISA section 8 (group0 128b, group1 256b).
// ---------------------------------------------------------------------------
__device__ inline void tdm_load_tile_f16(unsigned lds_addr,
                                         const _Float16* gsrc,
                                         unsigned rowStride, unsigned nrows,
                                         unsigned tile_w, unsigned tile_h) {
  unsigned long long ga = (unsigned long long)(size_t)gsrc;
  u32x4 g0;
  g0[0] = 1u;                                            // count=1, user mode
  g0[1] = lds_addr;                                      // LDS byte address
  g0[2] = (unsigned)(ga & 0xFFFFFFFFu);                  // global_addr[31:0]
  g0[3] = (unsigned)((ga >> 32) & 0x01FFFFFFu) | (2u << 30);  // [56:32]|type=2

  // group1: data_size=2B(1), pad_enable, pad_interval=16dw(3), pad_amount=2dw(1)
  unsigned long long q0 = (unsigned long long)((1u << 16) | (1u << 20) |
                                               (3u << 22) | (1u << 25))
                        | ((unsigned long long)(rowStride & 0xFFFFu) << 48);
  unsigned long long q1 = (unsigned long long)(rowStride >> 16)
                        | ((unsigned long long)nrows << 16)
                        | ((unsigned long long)tile_w << 48);
  unsigned long long q2 = (unsigned long long)tile_h
                        | ((unsigned long long)rowStride << 32);  // dim0 stride
  unsigned long long q3 = 0ull;
  i32x8 g1;
  g1[0] = (int)q0; g1[1] = (int)(q0 >> 32);
  g1[2] = (int)q1; g1[3] = (int)(q1 >> 32);
  g1[4] = (int)q2; g1[5] = (int)(q2 >> 32);
  g1[6] = (int)q3; g1[7] = (int)(q3 >> 32);
  i32x4 z4 = {0, 0, 0, 0};
  i32x8 z8 = {0, 0, 0, 0, 0, 0, 0, 0};
  // clang-23 / therock header form: 6 args (extra descriptor group + cpol)
  __builtin_amdgcn_tensor_load_to_lds(g0, g1, z4, z4, z8, 0);
}

// ---------------------------------------------------------------------------
// WMMA GEMM: C[m,n] = sum_k A16[m,k]*W16[n,k] (+bias) (+resid); outputs f32
// C and/or f16 C16. Block 256 = 8 waves, C tile 128x128, BK=32. Tiles DMA'd
// into double-buffered LDS by the Tensor Data Mover (wave 0 issues, TENSORcnt
// + workgroup barrier synchronize). Each wave: 2x4 16x16 tiles -> 8 WMMAs/step.
// Requires M%128==0, N%128==0, K%32==0.
// ---------------------------------------------------------------------------
#define BM_ 128
#define BN_ 128
#define BK_ 32

__global__ __launch_bounds__(256)
void gemm_wmma_kernel(const _Float16* __restrict__ A16,
                      const _Float16* __restrict__ W16,
                      const float* __restrict__ bias, const float* resid,
                      float* C, _Float16* C16, int M, int N, int K) {
  __shared__ _Float16 As[2][BM_][36];
  __shared__ _Float16 Bs[2][BN_][36];
  const int m0 = blockIdx.y * BM_, n0 = blockIdx.x * BN_;
  const int tid = threadIdx.x;
  const int wav = tid >> 5, lane = tid & 31;
  const int half16 = (lane >> 4) & 1, l16 = lane & 15;
  const int mg = wav & 3;        // 2 m-tiles: rows [mg*32, mg*32+32)
  const int ng = wav >> 2;       // 4 n-tiles: cols [ng*64, ng*64+64)

  v8f acc[2][4] = {};
  const int nsteps = K / BK_;

  auto issue = [&](int buf, int k0) {
    tdm_load_tile_f16((unsigned)(size_t)&As[buf][0][0],
                      A16 + (size_t)m0 * K + k0, (unsigned)K,
                      (unsigned)(M - m0), BK_, BM_);
    tdm_load_tile_f16((unsigned)(size_t)&Bs[buf][0][0],
                      W16 + (size_t)n0 * K + k0, (unsigned)K,
                      (unsigned)(N - n0), BK_, BN_);
  };

  if (wav == 0) issue(0, 0);
  for (int st = 0; st < nsteps; ++st) {
    if (wav == 0) __builtin_amdgcn_s_wait_tensorcnt(0);  // buf st&1 complete
    __syncthreads();               // all waves: data visible, prev reads done
    if (wav == 0 && st + 1 < nsteps) issue((st + 1) & 1, (st + 1) * BK_);
    const int buf = st & 1;

    v16h af[2], bf[4];
    #pragma unroll
    for (int t = 0; t < 2; ++t) {
      int row = (mg * 2 + t) * 16 + l16;
      #pragma unroll
      for (int j = 0; j < 16; ++j) {
        int kk = (j < 8 ? j : j + 8) + half16 * 8;  // CDNA5 16-bit A 16x32
        af[t][j] = As[buf][row][kk];
      }
    }
    #pragma unroll
    for (int t = 0; t < 4; ++t) {
      int col = (ng * 4 + t) * 16 + l16;
      #pragma unroll
      for (int j = 0; j < 16; ++j) {
        int kk = (j < 8 ? j : j + 8) + half16 * 8;  // CDNA5 16-bit B 32x16
        bf[t][j] = Bs[buf][col][kk];
      }
    }
    #pragma unroll
    for (int i = 0; i < 2; ++i)
      #pragma unroll
      for (int t = 0; t < 4; ++t)
        acc[i][t] = __builtin_amdgcn_wmma_f32_16x16x32_f16(
            false, af[i], false, bf[t], (short)0, acc[i][t], false, false);
  }

  #pragma unroll
  for (int i = 0; i < 2; ++i) {
    #pragma unroll
    for (int t = 0; t < 4; ++t) {
      #pragma unroll
      for (int r = 0; r < 8; ++r) {
        int row = m0 + (mg * 2 + i) * 16 + r + half16 * 8;  // CDNA5 f32 C/D
        int col = n0 + (ng * 4 + t) * 16 + l16;
        float v = acc[i][t][r];
        if (bias)  v += bias[col];
        if (resid) v += resid[(size_t)row * N + col];
        if (C)     C[(size_t)row * N + col] = v;
        if (C16)   C16[(size_t)row * N + col] = (_Float16)v;
      }
    }
  }
}

// ---------------------------------------------------------------------------
// Causal flash attention for one (b,h): f16 Q,K,V in [B,S,D], head h in cols
// [h*64,h*64+64); f16 output. Block 128 = 4 waves; each wave owns 16 q rows,
// processes 32 k-cols/iter (two score tiles -> full 16x32 P fragment).
// Online softmax via shfl_xor in 16-lane halves; P through per-wave LDS
// guarded by s_wait_dscnt (waves have different causal trip counts).
// ---------------------------------------------------------------------------
__global__ __launch_bounds__(128)
void flash_attn_kernel(const _Float16* __restrict__ Q,
                       const _Float16* __restrict__ Km,
                       const _Float16* __restrict__ Vm,
                       _Float16* __restrict__ O) {
  const int b = blockIdx.z, h = blockIdx.y;
  const int wav = threadIdx.x >> 5, lane = threadIdx.x & 31;
  const int half16 = (lane >> 4) & 1, l16 = lane & 15;
  const int q0 = blockIdx.x * 64 + wav * 16;
  const int hbase = h * DK_;
  const _Float16* Qb = Q  + (size_t)b * S_ * D_;
  const _Float16* Kb = Km + (size_t)b * S_ * D_;
  const _Float16* Vb = Vm + (size_t)b * S_ * D_;
  _Float16*       Ob = O  + (size_t)b * S_ * D_;

  __shared__ _Float16 Pl[4][16][36];

  v16h aq0, aq1;
  {
    int row = q0 + l16;
    #pragma unroll
    for (int j = 0; j < 16; ++j) {
      int kk = (j < 8 ? j : j + 8) + half16 * 8;
      aq0[j] = Qb[(size_t)row * D_ + hbase + kk];
      aq1[j] = Qb[(size_t)row * D_ + hbase + 32 + kk];
    }
  }

  v8f o0 = {}, o1 = {}, o2 = {}, o3 = {};
  float mrow[8], lrow[8];
  #pragma unroll
  for (int r = 0; r < 8; ++r) { mrow[r] = -3.0e38f; lrow[r] = 0.0f; }

  const int nblk = (q0 + 47) >> 5;
  for (int kb = 0; kb < nblk; ++kb) {
    const int kbase = kb * 32;

    v16h bk00, bk01, bk10, bk11;
    {
      const _Float16* kr0 = Kb + (size_t)(kbase + l16) * D_ + hbase;
      const _Float16* kr1 = Kb + (size_t)(kbase + 16 + l16) * D_ + hbase;
      #pragma unroll
      for (int j = 0; j < 16; ++j) {
        int dd = (j < 8 ? j : j + 8) + half16 * 8;
        bk00[j] = kr0[dd];
        bk01[j] = kr0[32 + dd];
        bk10[j] = kr1[dd];
        bk11[j] = kr1[32 + dd];
      }
    }
    v8f s0 = {}, s1 = {};
    s0 = __builtin_amdgcn_wmma_f32_16x16x32_f16(false, aq0, false, bk00,
                                                (short)0, s0, false, false);
    s0 = __builtin_amdgcn_wmma_f32_16x16x32_f16(false, aq1, false, bk01,
                                                (short)0, s0, false, false);
    s1 = __builtin_amdgcn_wmma_f32_16x16x32_f16(false, aq0, false, bk10,
                                                (short)0, s1, false, false);
    s1 = __builtin_amdgcn_wmma_f32_16x16x32_f16(false, aq1, false, bk11,
                                                (short)0, s1, false, false);

    const bool needMask = (kbase + 31 > q0);
    #pragma unroll
    for (int r = 0; r < 8; ++r) {
      float s0v = s0[r] * 0.125f;
      float s1v = s1[r] * 0.125f;
      if (needMask) {
        int qrow = q0 + r + half16 * 8;
        if (kbase + l16      > qrow) s0v = -1.0e30f;
        if (kbase + 16 + l16 > qrow) s1v = -1.0e30f;
      }
      float rm = fmaxf(s0v, s1v);
      rm = fmaxf(rm, __shfl_xor(rm, 1, 32));
      rm = fmaxf(rm, __shfl_xor(rm, 2, 32));
      rm = fmaxf(rm, __shfl_xor(rm, 4, 32));
      rm = fmaxf(rm, __shfl_xor(rm, 8, 32));
      float mnew  = fmaxf(mrow[r], rm);
      float scale = __expf(mrow[r] - mnew);
      float p0 = __expf(s0v - mnew);
      float p1 = __expf(s1v - mnew);
      float ps = p0 + p1;
      ps += __shfl_xor(ps, 1, 32);
      ps += __shfl_xor(ps, 2, 32);
      ps += __shfl_xor(ps, 4, 32);
      ps += __shfl_xor(ps, 8, 32);
      lrow[r] = lrow[r] * scale + ps;
      mrow[r] = mnew;
      o0[r] *= scale; o1[r] *= scale; o2[r] *= scale; o3[r] *= scale;
      int rr = r + half16 * 8;
      Pl[wav][rr][l16]      = (_Float16)p0;
      Pl[wav][rr][16 + l16] = (_Float16)p1;
    }
    asm volatile("s_wait_dscnt 0" ::: "memory");

    v16h ap;
    #pragma unroll
    for (int j = 0; j < 16; ++j) {
      int kk = (j < 8 ? j : j + 8) + half16 * 8;
      ap[j] = Pl[wav][l16][kk];
    }
    #pragma unroll
    for (int g = 0; g < 4; ++g) {
      v16h bv;
      #pragma unroll
      for (int j = 0; j < 16; ++j) {
        int kk = (j < 8 ? j : j + 8) + half16 * 8;
        bv[j] = Vb[(size_t)(kbase + kk) * D_ + hbase + g * 16 + l16];
      }
      v8f* op = (g == 0) ? &o0 : (g == 1) ? &o1 : (g == 2) ? &o2 : &o3;
      *op = __builtin_amdgcn_wmma_f32_16x16x32_f16(false, ap, false, bv,
                                                   (short)0, *op, false, false);
    }
    asm volatile("s_wait_dscnt 0" ::: "memory");
  }

  #pragma unroll
  for (int r = 0; r < 8; ++r) {
    int row = q0 + r + half16 * 8;
    float inv = 1.0f / lrow[r];
    Ob[(size_t)row * D_ + hbase + 0  + l16] = (_Float16)(o0[r] * inv);
    Ob[(size_t)row * D_ + hbase + 16 + l16] = (_Float16)(o1[r] * inv);
    Ob[(size_t)row * D_ + hbase + 32 + l16] = (_Float16)(o2[r] * inv);
    Ob[(size_t)row * D_ + hbase + 48 + l16] = (_Float16)(o3[r] * inv);
  }
}

// ---------------------------------------------------------------------------
// Host-side orchestration (all on `stream`; graph-capture safe)
// ---------------------------------------------------------------------------
extern "C" void kernel_launch(void* const* d_in, const int* in_sizes, int n_in,
                              void* d_out, int out_size, void* d_ws, size_t ws_size,
                              hipStream_t stream) {
  (void)in_sizes; (void)n_in; (void)out_size; (void)ws_size;

  const float* spectra = (const float*)d_in[0];
  const int*   stacks  = (const int*)  d_in[1];
  const float* fc_w    = (const float*)d_in[2];
  const float* fc_b    = (const float*)d_in[3];
  const float* emb     = (const float*)d_in[4];
  const float* sa_w    = (const float*)d_in[5];
  const float* sa_b    = (const float*)d_in[6];
  const float* ca_w    = (const float*)d_in[7];
  const float* ca_b    = (const float*)d_in[8];
  const float* ff_w1   = (const float*)d_in[9];
  const float* ff_b1   = (const float*)d_in[10];
  const float* ff_w2   = (const float*)d_in[11];
  const float* ff_b2   = (const float*)d_in[12];
  const float* ln_a    = (const float*)d_in[13];
  const float* ln_b    = (const float*)d_in[14];
  const float* fin_a   = (const float*)d_in[15];
  const float* fin_b   = (const float*)d_in[16];
  const float* gen_w   = (const float*)d_in[17];
  const float* gen_b   = (const float*)d_in[18];
  float* out = (float*)d_out;

  // workspace carve (~116 MB)
  char* ws = (char*)d_ws;
  size_t off = 0;
  auto carve = [&](size_t bytes) -> void* {
    void* p = (void*)(ws + off);
    off += (bytes + 255) & ~(size_t)255;
    return p;
  };
  const size_t MRxD2 = (size_t)B_ * S_ * D_ * sizeof(_Float16);
  float*    memv = (float*)carve((size_t)B_ * D_ * sizeof(float));
  float*    x    = (float*)carve((size_t)B_ * S_ * D_ * sizeof(float));
  _Float16* h16  = (_Float16*)carve(MRxD2);
  _Float16* q16  = (_Float16*)carve(MRxD2);
  _Float16* k16  = (_Float16*)carve(MRxD2);
  _Float16* v16  = (_Float16*)carve(MRxD2);
  _Float16* a16  = (_Float16*)carve(MRxD2);
  _Float16* f16b = (_Float16*)carve((size_t)B_ * S_ * DFF_ * sizeof(_Float16));
  float*    tmpv = (float*)carve((size_t)B_ * D_ * sizeof(float));
  float*    cav  = (float*)carve((size_t)B_ * D_ * sizeof(float));
  _Float16* saw16 = (_Float16*)carve((size_t)L_ * 4 * D_ * D_ * sizeof(_Float16));
  _Float16* fw116 = (_Float16*)carve((size_t)L_ * DFF_ * D_ * sizeof(_Float16));
  _Float16* fw216 = (_Float16*)carve((size_t)L_ * D_ * DFF_ * sizeof(_Float16));
  _Float16* gw16  = (_Float16*)carve((size_t)V_ * D_ * sizeof(_Float16));

  const int MR = B_ * S_;                 // 4096 rows
  const int nElems = B_ * S_ * D_;

  // one-time (per launch) weight conversion to f16
  {
    long long n;
    n = (long long)L_ * 4 * D_ * D_ / 4;
    cvt_f16_kernel<<<(unsigned)((n + 255) / 256), 256, 0, stream>>>(sa_w, saw16, n);
    n = (long long)L_ * DFF_ * D_ / 4;
    cvt_f16_kernel<<<(unsigned)((n + 255) / 256), 256, 0, stream>>>(ff_w1, fw116, n);
    n = (long long)L_ * D_ * DFF_ / 4;
    cvt_f16_kernel<<<(unsigned)((n + 255) / 256), 256, 0, stream>>>(ff_w2, fw216, n);
    n = (long long)V_ * D_ / 4;
    cvt_f16_kernel<<<(unsigned)((n + 255) / 256), 256, 0, stream>>>(gen_w, gw16, n);
  }

  // memory = fc(flatten(spectra))  [B,D];  x = emb[tgt_in]*sqrt(D)+PE
  gemv_kernel<<<dim3(D_ / 64, B_), 64, 0, stream>>>(spectra, fc_w, fc_b, memv,
                                                    FLAT_, D_);
  embed_kernel<<<(nElems + 255) / 256, 256, 0, stream>>>(stacks, emb, x);

  for (int l = 0; l < L_; ++l) {
    const _Float16* saw = saw16 + (size_t)l * 4 * D_ * D_;
    const float*    sab = sa_b  + (size_t)l * 4 * D_;
    const float*    caw = ca_w  + (size_t)l * 4 * D_ * D_;
    const float*    cab = ca_b  + (size_t)l * 4 * D_;
    const _Float16* w1  = fw116 + (size_t)l * DFF_ * D_;
    const float*    b1  = ff_b1 + (size_t)l * DFF_;
    const _Float16* w2  = fw216 + (size_t)l * D_ * DFF_;
    const float*    b2  = ff_b2 + (size_t)l * D_;
    const float*    la  = ln_a + (size_t)l * 3 * D_;
    const float*    lb  = ln_b + (size_t)l * 3 * D_;

    // --- self attention (pre-LN, causal) ---
    ln_kernel<<<MR, 256, 0, stream>>>(x, la + 0 * D_, lb + 0 * D_, h16);
    gemm_wmma_kernel<<<dim3(D_ / BN_, MR / BM_), 256, 0, stream>>>(
        h16, saw + 0 * (size_t)D_ * D_, sab + 0 * D_, nullptr,
        nullptr, q16, MR, D_, D_);
    gemm_wmma_kernel<<<dim3(D_ / BN_, MR / BM_), 256, 0, stream>>>(
        h16, saw + 1 * (size_t)D_ * D_, sab + 1 * D_, nullptr,
        nullptr, k16, MR, D_, D_);
    gemm_wmma_kernel<<<dim3(D_ / BN_, MR / BM_), 256, 0, stream>>>(
        h16, saw + 2 * (size_t)D_ * D_, sab + 2 * D_, nullptr,
        nullptr, v16, MR, D_, D_);
    flash_attn_kernel<<<dim3(S_ / 64, H_, B_), 128, 0, stream>>>(q16, k16, v16, a16);
    gemm_wmma_kernel<<<dim3(D_ / BN_, MR / BM_), 256, 0, stream>>>(
        a16, saw + 3 * (size_t)D_ * D_, sab + 3 * D_, x, x, nullptr, MR, D_, D_);

    // --- cross attention: Sk==1 => softmax==1 => q path dead code ---
    gemv_kernel<<<dim3(D_ / 64, B_), 64, 0, stream>>>(
        memv, caw + 2 * (size_t)D_ * D_, cab + 2 * D_, tmpv, D_, D_);
    gemv_kernel<<<dim3(D_ / 64, B_), 64, 0, stream>>>(
        tmpv, caw + 3 * (size_t)D_ * D_, cab + 3 * D_, cav, D_, D_);
    addvec_kernel<<<(nElems + 255) / 256, 256, 0, stream>>>(x, cav);

    // --- feed forward (no activation) ---
    ln_kernel<<<MR, 256, 0, stream>>>(x, la + 2 * D_, lb + 2 * D_, h16);
    gemm_wmma_kernel<<<dim3(DFF_ / BN_, MR / BM_), 256, 0, stream>>>(
        h16, w1, b1, nullptr, nullptr, f16b, MR, DFF_, D_);
    gemm_wmma_kernel<<<dim3(D_ / BN_, MR / BM_), 256, 0, stream>>>(
        f16b, w2, b2, x, x, nullptr, MR, D_, DFF_);
  }

  // final LN + generator projection (4096 x 32000 x 512)
  ln_kernel<<<MR, 256, 0, stream>>>(x, fin_a, fin_b, h16);
  gemm_wmma_kernel<<<dim3(V_ / BN_, MR / BM_), 256, 0, stream>>>(
      h16, gw16, gen_b, nullptr, out, nullptr, MR, V_, D_);
}
